// SelfAttentionBlock_47510928228767
// MI455X (gfx1250) — matrix operand order repaired and checked by reading, not compiled, not run on hardware
//
#include <hip/hip_runtime.h>
#include <hip/hip_bf16.h>
#include <math.h>

// ---------------------------------------------------------------------------
// SelfAttentionBlock for MI455X (gfx1250): bf16 WMMA GEMM/conv/attention
// ---------------------------------------------------------------------------

typedef unsigned short u16;
typedef __bf16 bf16;
typedef bf16 v16bf __attribute__((ext_vector_type(16)));
typedef float v8f   __attribute__((ext_vector_type(8)));

#define CC   192
#define HWIM 128
#define HW   16384          // 128*128
#define PIX  32768          // B*H*W
#define NB   2

__device__ __forceinline__ u16 f2bf(float f) {
  unsigned u = __float_as_uint(f);
  u = (u + 0x7fffu + ((u >> 16) & 1u)) >> 16;
  return (u16)u;
}
__device__ __forceinline__ bf16 us2bf(u16 s) {
  union { u16 s; bf16 b; } cv; cv.s = s; return cv.b;
}

__device__ __forceinline__ v8f wmma_bf16(v16bf a, v16bf b, v8f c) {
  return __builtin_amdgcn_wmma_f32_16x16x32_bf16(
      false, a, false, b, (short)0, c, false, false);
}

template <int ACT>
__device__ __forceinline__ float act_apply(float x, float a) {
  if (ACT == 1) return fmaxf(x, 0.f);                               // ReLU
  if (ACT == 2) return x >= 0.f ? x : a * x;                        // PReLU
  if (ACT == 3) return 0.5f * x * (1.f + erff(x * 0.70710678118654752f)); // GELU
  return x;
}

// ---------------------------------------------------------------------------
// Layout transposes (NCHW <-> NHWC), LDS tiled
// ---------------------------------------------------------------------------
__global__ __launch_bounds__(256) void k_nchw2nhwc(const float* __restrict__ in,
                                                   float* __restrict__ out) {
  __shared__ float t[32][33];
  int hw0 = blockIdx.x * 32, c0 = blockIdx.y * 32, b = blockIdx.z;
  int tx = threadIdx.x, ty = threadIdx.y;
  for (int i = 0; i < 32; i += 8)
    t[ty + i][tx] = in[((size_t)(b * CC + c0 + ty + i)) * HW + hw0 + tx];
  __syncthreads();
  for (int i = 0; i < 32; i += 8)
    out[((size_t)b * HW + hw0 + ty + i) * CC + c0 + tx] = t[tx][ty + i];
}

__global__ __launch_bounds__(256) void k_nhwc2nchw(const float* __restrict__ in,
                                                   float* __restrict__ out) {
  __shared__ float t[32][33];
  int hw0 = blockIdx.x * 32, c0 = blockIdx.y * 32, b = blockIdx.z;
  int tx = threadIdx.x, ty = threadIdx.y;
  for (int i = 0; i < 32; i += 8)
    t[ty + i][tx] = in[((size_t)b * HW + hw0 + ty + i) * CC + c0 + tx];
  __syncthreads();
  for (int i = 0; i < 32; i += 8)
    out[((size_t)(b * CC + c0 + ty + i)) * HW + hw0 + tx] = t[tx][ty + i];
}

// ---------------------------------------------------------------------------
// LayerNorm over channels (C=192), one wave32 per pixel, 6 values per lane
// ---------------------------------------------------------------------------
__global__ __launch_bounds__(256) void k_ln(const float* __restrict__ in,
                                            const float* __restrict__ g,
                                            const float* __restrict__ b,
                                            float* __restrict__ outf,
                                            u16* __restrict__ outb, int ldob) {
  int lane = threadIdx.x & 31;
  int pix = blockIdx.x * 8 + (threadIdx.x >> 5);
  const float* row = in + (size_t)pix * CC;
  float v[6], s = 0.f, sq = 0.f;
#pragma unroll
  for (int j = 0; j < 6; ++j) {
    v[j] = row[lane + 32 * j];
    s += v[j];
    sq += v[j] * v[j];
  }
#pragma unroll
  for (int m = 1; m < 32; m <<= 1) {
    s += __shfl_xor(s, m, 32);
    sq += __shfl_xor(sq, m, 32);
  }
  float mean = s * (1.f / 192.f);
  float var = sq * (1.f / 192.f) - mean * mean;
  float rstd = rsqrtf(var + 1e-5f);
#pragma unroll
  for (int j = 0; j < 6; ++j) {
    int c = lane + 32 * j;
    float o = (v[j] - mean) * rstd * g[c] + b[c];
    if (outf) outf[(size_t)pix * CC + c] = o;
    outb[(size_t)pix * ldob + c] = f2bf(o);
  }
}

// ---------------------------------------------------------------------------
// Weight packing: OI(11) -> [Cin][Cout] bf16 ; OI33 -> [tap*Cin+ci][Cout] bf16
// ---------------------------------------------------------------------------
__global__ void k_pack1(const float* __restrict__ w, u16* __restrict__ o,
                        int Cout, int Cin) {
  int i = blockIdx.x * 256 + threadIdx.x;
  if (i >= Cout * Cin) return;
  int co = i / Cin, ci = i - co * Cin;
  o[(size_t)ci * Cout + co] = f2bf(w[i]);
}
__global__ void k_pack3(const float* __restrict__ w, u16* __restrict__ o,
                        int Cout, int Cin) {
  int i = blockIdx.x * 256 + threadIdx.x;
  if (i >= Cout * Cin * 9) return;
  int co = i / (Cin * 9);
  int r = i - co * Cin * 9;
  int ci = r / 9, tap = r - ci * 9;
  o[((size_t)tap * Cin + ci) * Cout + co] = f2bf(w[i]);
}

// ---------------------------------------------------------------------------
// Dense relative-position-bias table: bt[h][n][m] = rpb[rpi[n*256+m]*6 + h]
// (1.5 MB, L2-resident; removes two-level gather from the attention loop)
// ---------------------------------------------------------------------------
__global__ __launch_bounds__(256) void k_bias(const float* __restrict__ rpb,
                                              const int* __restrict__ rpi,
                                              float* __restrict__ bt) {
  int i = blockIdx.x * 256 + threadIdx.x;     // 0 .. 65535
  int idx = rpi[i];
#pragma unroll
  for (int h = 0; h < 6; ++h)
    bt[(size_t)h * 65536 + i] = rpb[(size_t)idx * 6 + h];
}

// ---------------------------------------------------------------------------
// Generic WMMA GEMM: out[M,N] = act(A[M,K]*Bw[K,N] + bias) (+res), bf16 in.
// 256 threads / 8 waves, 128x64 block tile, 32x32 per wave, K step 32.
// ---------------------------------------------------------------------------
template <int ACT>
__global__ __launch_bounds__(256) void k_gemm(
    const u16* __restrict__ A, int lda, const u16* __restrict__ Bw,
    const float* __restrict__ bias, const float* __restrict__ alpha,
    const float* __restrict__ res, float* __restrict__ outf, int ldof,
    u16* __restrict__ outb, int ldob, int M, int N, int K) {
  (void)M;
  __shared__ u16 As[128 * 40];  // 80B row stride: 16B-aligned, conflict-spread
  __shared__ u16 Bs[64 * 40];   // stored transposed: Bs[n][k]
  const int tid = threadIdx.x;
  const int lane = tid & 31;
  const int wid = tid >> 5;
  const int wm = wid >> 1, wn = wid & 1;      // 4 x 2 waves
  const int lc = lane & 15;
  const int half8 = (lane & 16) ? 8 : 0;
  const int half16 = (lane & 16) ? 16 : 0;
  const int m0 = blockIdx.x * 128;
  const int n0 = blockIdx.y * 64;

  v8f acc[2][2] = {};

  for (int kk = 0; kk < K; kk += 32) {
    __syncthreads();
    // stage A (128 x 32), uint4 chunks of 8 bf16, zero-fill K tail
    for (int i = tid; i < 512; i += 256) {
      int m = i >> 2;
      int c8 = (i & 3) << 3;
      int k = kk + c8;
      uint4 v = make_uint4(0, 0, 0, 0);
      if (k < K) v = *(const uint4*)(A + (size_t)(m0 + m) * lda + k);
      *(uint4*)(&As[m * 40 + c8]) = v;
    }
    // stage B transposed (Bs[n][k]), zero-fill K/N tails
    {
      int i = tid;  // 256 chunks, one per thread
      int kr = i >> 3;
      int n8 = (i & 7) << 3;
      int k = kk + kr;
      int n = n0 + n8;
      uint4 v = make_uint4(0, 0, 0, 0);
      if (k < K && n < N) v = *(const uint4*)(Bw + (size_t)k * N + n);
      const u16* vs = (const u16*)&v;
#pragma unroll
      for (int j = 0; j < 8; ++j) Bs[(n8 + j) * 40 + kr] = vs[j];
    }
    if (kk + 32 < K)
      __builtin_prefetch(A + (size_t)(m0 + (tid & 127)) * lda + kk + 32, 0, 1);
    __syncthreads();

    v16bf af[2], bfr[2];
#pragma unroll
    for (int i = 0; i < 2; ++i) {
      const u16* ar = &As[(wm * 32 + i * 16 + lc) * 40];
#pragma unroll
      for (int j = 0; j < 8; ++j) {
        af[i][j] = us2bf(ar[half16 + j]);
        af[i][8 + j] = us2bf(ar[half16 + 8 + j]);
      }
    }
#pragma unroll
    for (int jn = 0; jn < 2; ++jn) {
      const u16* br = &Bs[(wn * 32 + jn * 16 + lc) * 40];
#pragma unroll
      for (int j = 0; j < 16; ++j) bfr[jn][j] = us2bf(br[half16 + j]);
    }
#pragma unroll
    for (int i = 0; i < 2; ++i)
#pragma unroll
      for (int jn = 0; jn < 2; ++jn)
        acc[i][jn] = wmma_bf16(af[i], bfr[jn], acc[i][jn]);
  }

  const float aV = (ACT == 2 && alpha) ? alpha[0] : 0.f;
#pragma unroll
  for (int i = 0; i < 2; ++i)
#pragma unroll
    for (int jn = 0; jn < 2; ++jn)
#pragma unroll
      for (int r = 0; r < 8; ++r) {
        int row = m0 + wm * 32 + i * 16 + r + half8;
        int col = n0 + wn * 32 + jn * 16 + lc;
        if (col < N) {
          float v = acc[i][jn][r] + bias[col];
          v = act_apply<ACT>(v, aV);
          if (res) v += res[(size_t)row * N + col];
          if (outf) outf[(size_t)row * ldof + col] = v;
          if (outb) outb[(size_t)row * ldob + col] = f2bf(v);
        }
      }
}

// ---------------------------------------------------------------------------
// 3x3 conv (pad=1) as implicit GEMM with WMMA: K = 9*192 = 1728, N = 192.
// A tile gathered from NHWC bf16 input with zero padding; 128-pixel M tile
// is exactly one image row (W=128).
// ---------------------------------------------------------------------------
template <int ACT>
__global__ __launch_bounds__(256) void k_conv3(
    const u16* __restrict__ In, int lda, const u16* __restrict__ Bw,
    const float* __restrict__ bias, const float* __restrict__ alpha,
    const float* __restrict__ res, float* __restrict__ outf, int ldof,
    u16* __restrict__ outb, int ldob) {
  const int N = 192, K = 1728;
  __shared__ u16 As[128 * 40];
  __shared__ u16 Bs[64 * 40];
  const int tid = threadIdx.x;
  const int lane = tid & 31;
  const int wid = tid >> 5;
  const int wm = wid >> 1, wn = wid & 1;
  const int lc = lane & 15;
  const int half8 = (lane & 16) ? 8 : 0;
  const int half16 = (lane & 16) ? 16 : 0;
  const int m0 = blockIdx.x * 128;
  const int n0 = blockIdx.y * 64;

  v8f acc[2][2] = {};

  for (int kk = 0; kk < K; kk += 32) {
    int tap = kk / CC;            // one tap per K-step (192 % 32 == 0)
    int c0 = kk - tap * CC;
    int dy = tap / 3 - 1, dx = tap % 3 - 1;
    __syncthreads();
    for (int i = tid; i < 512; i += 256) {
      int m = i >> 2;
      int c8 = (i & 3) << 3;
      int p = m0 + m;
      int b = p >> 14, hw = p & 16383;
      int h = (hw >> 7) + dy, w = (hw & 127) + dx;
      uint4 v = make_uint4(0, 0, 0, 0);
      if ((unsigned)h < 128u && (unsigned)w < 128u)
        v = *(const uint4*)(In + ((size_t)b * HW + h * 128 + w) * lda + c0 + c8);
      *(uint4*)(&As[m * 40 + c8]) = v;
    }
    {
      int i = tid;
      int kr = i >> 3;
      int n8 = (i & 7) << 3;
      uint4 v = *(const uint4*)(Bw + (size_t)(kk + kr) * N + n0 + n8);
      const u16* vs = (const u16*)&v;
#pragma unroll
      for (int j = 0; j < 8; ++j) Bs[(n8 + j) * 40 + kr] = vs[j];
    }
    __syncthreads();

    v16bf af[2], bfr[2];
#pragma unroll
    for (int i = 0; i < 2; ++i) {
      const u16* ar = &As[(wm * 32 + i * 16 + lc) * 40];
#pragma unroll
      for (int j = 0; j < 8; ++j) {
        af[i][j] = us2bf(ar[half16 + j]);
        af[i][8 + j] = us2bf(ar[half16 + 8 + j]);
      }
    }
#pragma unroll
    for (int jn = 0; jn < 2; ++jn) {
      const u16* br = &Bs[(wn * 32 + jn * 16 + lc) * 40];
#pragma unroll
      for (int j = 0; j < 16; ++j) bfr[jn][j] = us2bf(br[half16 + j]);
    }
#pragma unroll
    for (int i = 0; i < 2; ++i)
#pragma unroll
      for (int jn = 0; jn < 2; ++jn)
        acc[i][jn] = wmma_bf16(af[i], bfr[jn], acc[i][jn]);
  }

  const float aV = (ACT == 2 && alpha) ? alpha[0] : 0.f;
#pragma unroll
  for (int i = 0; i < 2; ++i)
#pragma unroll
    for (int jn = 0; jn < 2; ++jn)
#pragma unroll
      for (int r = 0; r < 8; ++r) {
        int row = m0 + wm * 32 + i * 16 + r + half8;
        int col = n0 + wn * 32 + jn * 16 + lc;
        float v = acc[i][jn][r] + bias[col];
        v = act_apply<ACT>(v, aV);
        if (res) v += res[(size_t)row * N + col];
        if (outf) outf[(size_t)row * ldof + col] = v;
        if (outb) outb[(size_t)row * ldob + col] = f2bf(v);
      }
}

// ---------------------------------------------------------------------------
// Involution aggregation: x_inv[p,c] = sum_k wgt[p, (c/2)*9+k] * xn[nbr_k, c]
// ---------------------------------------------------------------------------
__global__ __launch_bounds__(256) void k_involution(const float* __restrict__ xn,
                                                    const float* __restrict__ wgt,
                                                    u16* __restrict__ outb, int ldo) {
  int i = blockIdx.x * 256 + threadIdx.x;
  int p = i / CC, c = i - p * CC;
  int b = p >> 14, hw = p & 16383, h = hw >> 7, w = hw & 127;
  const float* wg = wgt + (size_t)p * 864 + (size_t)(c >> 1) * 9;
  float acc = 0.f;
#pragma unroll
  for (int k = 0; k < 9; ++k) {
    int h2 = h + k / 3 - 1, w2 = w + k % 3 - 1;
    if ((unsigned)h2 < 128u && (unsigned)w2 < 128u)
      acc += wg[k] * xn[((size_t)b * HW + h2 * 128 + w2) * CC + c];
  }
  outb[(size_t)p * ldo + c] = f2bf(acc);
}

// ---------------------------------------------------------------------------
// Window partition / reverse (16x16 windows, 8x8 grid per batch)
// ---------------------------------------------------------------------------
__global__ __launch_bounds__(256) void k_partition(const u16* __restrict__ xw,
                                                   u16* __restrict__ win) {
  int i = blockIdx.x * 256 + threadIdx.x;       // (token, 8-chan chunk)
  int t = i / 24, ch = (i - t * 24) * 8;
  int wi = t >> 8, n = t & 255;
  int b = wi >> 6, r = wi & 63, wh = r >> 3, ww = r & 7;
  int h = wh * 16 + (n >> 4), w = ww * 16 + (n & 15);
  size_t p = (size_t)b * HW + h * 128 + w;
  *(uint4*)(win + (size_t)t * CC + ch) = *(const uint4*)(xw + p * CC + ch);
}

__global__ __launch_bounds__(256) void k_reverse(const float* __restrict__ projf,
                                                 const float* __restrict__ shortcutf,
                                                 float* __restrict__ resid,
                                                 u16* __restrict__ xrb) {
  int i = blockIdx.x * 256 + threadIdx.x;
  int p = i / CC, c = i - p * CC;
  int b = p >> 14, hw = p & 16383, h = hw >> 7, w = hw & 127;
  int wi = b * 64 + (h >> 4) * 8 + (w >> 4);
  int t = wi * 256 + (h & 15) * 16 + (w & 15);
  float pv = projf[(size_t)t * CC + c];
  resid[(size_t)p * CC + c] = shortcutf[(size_t)p * CC + c] + pv;
  xrb[(size_t)p * CC + c] = f2bf(pv);
}

// ---------------------------------------------------------------------------
// Window attention: one block per (window, head). 4 waves, 16-row strips.
// S = Q*K^T via WMMA. Score tiles are not kept resident (no spills): pass 1
// computes the row max, pass 2 recomputes S, exponentiates, streams bf16 P
// to LDS and accumulates the row sum. Then O = P*V via WMMA. Bias comes from
// the dense per-head table (plain coalesced loads, no indirect gather).
// ---------------------------------------------------------------------------
__global__ __launch_bounds__(128) void k_attn(const u16* __restrict__ qkv,
                                              const float* __restrict__ biasT,
                                              u16* __restrict__ outb) {
  __shared__ u16 Klds[256 * 32];     // K[token][d]
  __shared__ u16 Vt[32 * 256];       // V^T[d][token]
  __shared__ u16 Pl[4 * 16 * 256];   // per-wave P strips (bf16)
  const int wi = blockIdx.x / 6;
  const int hh = blockIdx.x % 6;
  const u16* base = qkv + (size_t)wi * 256 * 576;
  const float* bt = biasT + (size_t)hh * 65536;   // [256][256] for this head
  const int tid = threadIdx.x;
  for (int i = tid; i < 256 * 32; i += 128) {
    int tok = i >> 5, d = i & 31;
    const u16* row = base + (size_t)tok * 576 + hh * 32;
    Klds[tok * 32 + d] = row[192 + d];
    Vt[d * 256 + tok] = row[384 + d];
  }
  __syncthreads();
  const int lane = tid & 31, wid = tid >> 5;
  const int lc = lane & 15;
  const int half8 = (lane & 16) ? 8 : 0;
  const int half16 = (lane & 16) ? 16 : 0;
  u16* Pw = Pl + wid * 16 * 256;
  const float scale = 0.17677669529663688f;   // 1/sqrt(32)

  for (int s = wid; s < 16; s += 4) {
    const int row0 = s * 16;
    v16bf qf;
    {
      const u16* qr = base + (size_t)(row0 + lc) * 576 + hh * 32;
#pragma unroll
      for (int j = 0; j < 8; ++j) {
        qf[j] = us2bf(qr[half16 + j]);
        qf[8 + j] = us2bf(qr[half16 + 8 + j]);
      }
    }
    // ---- pass 1: row max (score tiles are transient) ----
    float rmax[8];
#pragma unroll
    for (int r = 0; r < 8; ++r) rmax[r] = -3.0e38f;
    for (int t = 0; t < 16; ++t) {
      v16bf kf;
      const u16* kr = Klds + (t * 16 + lc) * 32 + half16;
#pragma unroll
      for (int j = 0; j < 16; ++j) kf[j] = us2bf(kr[j]);
      v8f z = {};
      v8f c = wmma_bf16(qf, kf, z);
      const int col = t * 16 + lc;
#pragma unroll
      for (int r = 0; r < 8; ++r) {
        int row = row0 + r + half8;
        float v = c[r] * scale + bt[row * 256 + col];
        rmax[r] = fmaxf(rmax[r], v);
      }
    }
#pragma unroll
    for (int r = 0; r < 8; ++r)
#pragma unroll
      for (int m = 1; m < 16; m <<= 1)
        rmax[r] = fmaxf(rmax[r], __shfl_xor(rmax[r], m, 32));
    // ---- pass 2: recompute S, exponentiate, stream bf16 P to LDS ----
    float rsum[8] = {0.f, 0.f, 0.f, 0.f, 0.f, 0.f, 0.f, 0.f};
    for (int t = 0; t < 16; ++t) {
      v16bf kf;
      const u16* kr = Klds + (t * 16 + lc) * 32 + half16;
#pragma unroll
      for (int j = 0; j < 16; ++j) kf[j] = us2bf(kr[j]);
      v8f z = {};
      v8f c = wmma_bf16(qf, kf, z);
      const int col = t * 16 + lc;
#pragma unroll
      for (int r = 0; r < 8; ++r) {
        int row = row0 + r + half8;
        float v = c[r] * scale + bt[row * 256 + col];
        float p = __expf(v - rmax[r]);
        rsum[r] += p;
        Pw[(r + half8) * 256 + t * 16 + lc] = f2bf(p);
      }
    }
#pragma unroll
    for (int r = 0; r < 8; ++r)
#pragma unroll
      for (int m = 1; m < 16; m <<= 1) rsum[r] += __shfl_xor(rsum[r], m, 32);
    __syncthreads();  // uniform: every wave runs exactly 4 strips
    // ---- O = P x V ----
    v8f O0 = {}, O1 = {};
    for (int ks = 0; ks < 8; ++ks) {
      v16bf pf;
      const u16* pr = Pw + lc * 256 + ks * 32;
#pragma unroll
      for (int j = 0; j < 8; ++j) {
        pf[j] = us2bf(pr[half16 + j]);
        pf[8 + j] = us2bf(pr[half16 + 8 + j]);
      }
      v16bf vf0, vf1;
      const u16* vr0 = Vt + lc * 256 + ks * 32 + half16;
      const u16* vr1 = Vt + (16 + lc) * 256 + ks * 32 + half16;
#pragma unroll
      for (int j = 0; j < 16; ++j) {
        vf0[j] = us2bf(vr0[j]);
        vf1[j] = us2bf(vr1[j]);
      }
      O0 = wmma_bf16(pf, vf0, O0);
      O1 = wmma_bf16(pf, vf1, O1);
    }
#pragma unroll
    for (int r = 0; r < 8; ++r) {
      int row = row0 + r + half8;
      float inv = 1.f / rsum[r];
      size_t ob = ((size_t)wi * 256 + row) * CC + hh * 32;
      outb[ob + lc] = f2bf(O0[r] * inv);
      outb[ob + 16 + lc] = f2bf(O1[r] * inv);
    }
  }
}

// ---------------------------------------------------------------------------
// Host-side orchestration
// ---------------------------------------------------------------------------
extern "C" void kernel_launch(void* const* d_in, const int* in_sizes, int n_in,
                              void* d_out, int out_size, void* d_ws, size_t ws_size,
                              hipStream_t stream) {
  (void)in_sizes; (void)n_in; (void)out_size; (void)ws_size;
  const float* x         = (const float*)d_in[0];
  const float* g1        = (const float*)d_in[1];
  const float* b1        = (const float*)d_in[2];
  const float* inv_c1_w  = (const float*)d_in[3];
  const float* inv_c1_b  = (const float*)d_in[4];
  const float* inv_prelu = (const float*)d_in[5];
  const float* inv_c2_w  = (const float*)d_in[6];
  const float* inv_c2_b  = (const float*)d_in[7];
  const float* con_w     = (const float*)d_in[8];
  const float* con_b     = (const float*)d_in[9];
  const float* f1_w      = (const float*)d_in[10];
  const float* f1_b      = (const float*)d_in[11];
  const float* f2_w      = (const float*)d_in[12];
  const float* f2_b      = (const float*)d_in[13];
  const float* qkv_w     = (const float*)d_in[14];
  const float* qkv_b     = (const float*)d_in[15];
  const float* rpb       = (const float*)d_in[16];
  const float* proj_w    = (const float*)d_in[17];
  const float* proj_b    = (const float*)d_in[18];
  const float* rev_w     = (const float*)d_in[19];
  const float* rev_b     = (const float*)d_in[20];
  const float* rev_prelu = (const float*)d_in[21];
  const float* g2        = (const float*)d_in[22];
  const float* b2        = (const float*)d_in[23];
  const float* fc1_w     = (const float*)d_in[24];
  const float* fc1_b     = (const float*)d_in[25];
  const float* fc2_w     = (const float*)d_in[26];
  const float* fc2_b     = (const float*)d_in[27];
  const int*   rpi       = (const int*)d_in[28];

  // ---- workspace arena (lifetime-based reuse) ----
  char* ws = (char*)d_ws;
  size_t off = 0;
  auto alloc = [&](size_t bytes) -> char* {
    char* p = ws + off;
    off += (bytes + 255) & ~(size_t)255;
    return p;
  };
  float* shortcutf = (float*)alloc((size_t)PIX * CC * 4);   // BHWC input (persist)
  float* yf        = (float*)alloc((size_t)PIX * CC * 4);   // y (persist)
  u16*   cat       = (u16*)  alloc((size_t)PIX * 576 * 2);  // [xn | x_inv | x_conv]
  float* xnf       = (float*)alloc((size_t)PIX * CC * 4);   // xn f32 -> proj -> final
  char*  big       =         alloc((size_t)PIX * 864 * 4);  // wgt -> qkv+hidden+resid
  u16*   t1        = (u16*)  alloc((size_t)PIX * 48 * 2);
  u16*   buf6      = (u16*)  alloc((size_t)PIX * CC * 2);   // fused1 -> attn -> xr_bf
  u16*   xw        = (u16*)  alloc((size_t)PIX * CC * 2);
  u16*   buf8      = (u16*)  alloc((size_t)PIX * CC * 2);   // win -> h_bf
  u16*   wpk       = (u16*)  alloc(1u << 21);               // packed weights (reused)
  float* biasT     = (float*)alloc((size_t)6 * 65536 * 4);  // dense RPB per head

  float* wgt   = (float*)big;
  u16*   qkvb  = (u16*)big;
  u16*   hhb   = (u16*)(big + (size_t)PIX * 576 * 2);
  float* resid = (float*)(big + (size_t)PIX * 576 * 2 + (size_t)PIX * 768 * 2);

  const dim3 tb(32, 8);
  const dim3 tg(HW / 32, CC / 32, NB);

  // 1. NCHW -> NHWC (shortcut)
  k_nchw2nhwc<<<tg, tb, 0, stream>>>(x, shortcutf);
  // 2. LN1 -> xn f32 + bf16 into cat[:,0:192]
  k_ln<<<PIX / 8, 256, 0, stream>>>(shortcutf, g1, b1, xnf, cat, 576);
  // 2b. dense relative-position-bias table
  k_bias<<<65536 / 256, 256, 0, stream>>>(rpb, rpi, biasT);
  // 3. involution weight branch: prelu(xn @ W1)  [P,48]
  k_pack1<<<(48 * 192 + 255) / 256, 256, 0, stream>>>(inv_c1_w, wpk, 48, 192);
  k_gemm<2><<<dim3(PIX / 128, 1), 256, 0, stream>>>(cat, 576, wpk, inv_c1_b, inv_prelu,
      nullptr, nullptr, 0, t1, 48, PIX, 48, 192);
  // 4. wgt = t1 @ W2  [P,864] f32
  k_pack1<<<(864 * 48 + 255) / 256, 256, 0, stream>>>(inv_c2_w, wpk, 864, 48);
  k_gemm<0><<<dim3(PIX / 128, 14), 256, 0, stream>>>(t1, 48, wpk, inv_c2_b, nullptr,
      nullptr, wgt, 864, nullptr, 0, PIX, 864, 48);
  // 5. involution aggregation -> cat[:,192:384]
  k_involution<<<PIX * CC / 256, 256, 0, stream>>>(xnf, wgt, cat + 192, 576);
  // 6. x_conv = conv3x3(x_inv) -> cat[:,384:576]
  k_pack3<<<(192 * 192 * 9 + 255) / 256, 256, 0, stream>>>(con_w, wpk, 192, 192);
  k_conv3<0><<<dim3(PIX / 128, 3), 256, 0, stream>>>(cat + 192, 576, wpk, con_b,
      nullptr, nullptr, nullptr, 0, cat + 384, 576);
  // 7. f1: relu(cat @ Wf1)  [P,192]
  k_pack1<<<(192 * 576 + 255) / 256, 256, 0, stream>>>(f1_w, wpk, 192, 576);
  k_gemm<1><<<dim3(PIX / 128, 3), 256, 0, stream>>>(cat, 576, wpk, f1_b, nullptr,
      nullptr, nullptr, 0, buf6, 192, PIX, 192, 576);
  // 8. f2: conv3x3 -> xw
  k_pack3<<<(192 * 192 * 9 + 255) / 256, 256, 0, stream>>>(f2_w, wpk, 192, 192);
  k_conv3<0><<<dim3(PIX / 128, 3), 256, 0, stream>>>(buf6, 192, wpk, f2_b, nullptr,
      nullptr, nullptr, 0, xw, 192);
  // 9. window partition -> win
  k_partition<<<PIX * 24 / 256, 256, 0, stream>>>(xw, buf8);
  // 10. qkv GEMM
  k_pack1<<<(576 * 192 + 255) / 256, 256, 0, stream>>>(qkv_w, wpk, 576, 192);
  k_gemm<0><<<dim3(PIX / 128, 9), 256, 0, stream>>>(buf8, 192, wpk, qkv_b, nullptr,
      nullptr, nullptr, 0, qkvb, 576, PIX, 576, 192);
  // 11. window attention
  k_attn<<<128 * 6, 128, 0, stream>>>(qkvb, biasT, buf6);
  // 12. proj
  k_pack1<<<(192 * 192 + 255) / 256, 256, 0, stream>>>(proj_w, wpk, 192, 192);
  k_gemm<0><<<dim3(PIX / 128, 3), 256, 0, stream>>>(buf6, 192, wpk, proj_b, nullptr,
      nullptr, xnf, 192, nullptr, 0, PIX, 192, 192);
  // 13. window reverse: resid = shortcut + xr ; xr bf16
  k_reverse<<<PIX * CC / 256, 256, 0, stream>>>(xnf, shortcutf, resid, buf6);
  // 14. rev conv: y = prelu(conv(xr)) + (shortcut + xr)
  k_pack3<<<(192 * 192 * 9 + 255) / 256, 256, 0, stream>>>(rev_w, wpk, 192, 192);
  k_conv3<2><<<dim3(PIX / 128, 3), 256, 0, stream>>>(buf6, 192, wpk, rev_b, rev_prelu,
      resid, yf, 192, nullptr, 0);
  // 15. LN2 -> h bf16
  k_ln<<<PIX / 8, 256, 0, stream>>>(yf, g2, b2, nullptr, buf8, 192);
  // 16. fc1 + exact GELU
  k_pack1<<<(768 * 192 + 255) / 256, 256, 0, stream>>>(fc1_w, wpk, 768, 192);
  k_gemm<3><<<dim3(PIX / 128, 12), 256, 0, stream>>>(buf8, 192, wpk, fc1_b, nullptr,
      nullptr, nullptr, 0, hhb, 768, PIX, 768, 192);
  // 17. fc2 + residual(y)
  k_pack1<<<(192 * 768 + 255) / 256, 256, 0, stream>>>(fc2_w, wpk, 192, 768);
  k_gemm<0><<<dim3(PIX / 128, 3), 256, 0, stream>>>(hhb, 768, wpk, fc2_b, nullptr,
      yf, xnf, 192, nullptr, 0, PIX, 192, 768);
  // 18. NHWC -> NCHW output
  k_nhwc2nchw<<<tg, tb, 0, stream>>>(xnf, (float*)d_out);
}
